// AttentionHead_53901839564972
// MI455X (gfx1250) — matrix operand order repaired and checked by reading, not compiled, not run on hardware
//
#include <hip/hip_runtime.h>
#include <hip/hip_bf16.h>

// ---------------------------------------------------------------------------
// AttentionHead for MI455X (gfx1250), wave32 + v_wmma_f32_16x16x32_f16.
//
// Stage 1: QKV projection GEMM (x[16384,1024] @ W^T[1024,64] + b), f16 WMMA,
//          f32 accum, double-buffered LDS W chunks, global loads for chunk
//          ec+1 issued before the 12 WMMAs of chunk ec. B fragments loaded in
//          two alternating groups of 4; amdgpu_waves_per_eu(1) relaxes the
//          scheduler's pressure target so both groups stay live and ds_load
//          latency overlaps the WMMA bursts instead of serializing.
//          sqrt(H)=8 softmax scale folded into Q. V stored transposed
//          [B,64,S] so stage 2 loads it as a contiguous A-operand.
// Stage 2: fused causal flash attention, transposed tiles:
//          S^T = K(16x64) @ Q^T(64x16)  -> query index on lane axis
//          O^T += V^T(16x32) @ P^T(32x16)
//          Double-buffered K/V LDS blocks; next block's global loads issued
//          before current block's 8 WMMAs; all 8 LDS fragments (K+V)
//          pre-loaded at iteration top; one barrier per iteration.
// ---------------------------------------------------------------------------

typedef __attribute__((ext_vector_type(16))) _Float16     v16h;
typedef __attribute__((ext_vector_type(8)))  float        v8f;
typedef __attribute__((ext_vector_type(4)))  float        v4f;
typedef __attribute__((ext_vector_type(4)))  unsigned int v4u;
typedef __attribute__((ext_vector_type(4)))  _Float16     v4h;

constexpr int BATCH = 8;
constexpr int SEQ   = 2048;
constexpr int EMB   = 1024;
constexpr int HD    = 64;

union V16HU { struct { v4u lo, hi; } u; v16h v; };

__device__ __forceinline__ v16h ld16_2(const _Float16* p0, const _Float16* p1) {
  V16HU t;
  t.u.lo = *(const v4u*)p0;   // 16B
  t.u.hi = *(const v4u*)p1;   // 16B
  return t.v;
}
__device__ __forceinline__ v16h ld16(const _Float16* p) { return ld16_2(p, p + 8); }

__device__ __forceinline__ v8f wmma_f16(v16h a, v16h b, v8f c) {
  return __builtin_amdgcn_wmma_f32_16x16x32_f16(false, a, false, b, (short)0, c,
                                                false, false);
}

__device__ __forceinline__ v16h cvt_a(const v4f xr[4]) {
  // A-fragment (16x32 f16): halves 0..7 <- xr[0..1], halves 8..15 <- xr[2..3]
  v16h a;
  #pragma unroll
  for (int i = 0; i < 4; ++i) {
    a[i]      = (_Float16)xr[0][i];
    a[4 + i]  = (_Float16)xr[1][i];
    a[8 + i]  = (_Float16)xr[2][i];
    a[12 + i] = (_Float16)xr[3][i];
  }
  return a;
}

// ---------------------------------------------------------------------------
// Stage 1: QKV projection. grid = (B*S)/128 blocks, 256 threads (8 waves).
// Each wave computes a 16-row x 64-col tile for Q, K and V (12 f32 C-frags).
// ---------------------------------------------------------------------------
__global__ __launch_bounds__(256)
__attribute__((amdgpu_waves_per_eu(1)))
void qkv_proj_kernel(
    const float* __restrict__ x,
    const float* __restrict__ Wk, const float* __restrict__ bk,
    const float* __restrict__ Wq, const float* __restrict__ bq,
    const float* __restrict__ Wv, const float* __restrict__ bv,
    _Float16* __restrict__ qbuf, _Float16* __restrict__ kbuf,
    _Float16* __restrict__ vtbuf)
{
  __shared__ __align__(32) _Float16 ldsW[2][3][64][32];   // 24 KB double buffer
  const int tid  = threadIdx.x;
  const int wave = tid >> 5;
  const int ln   = tid & 31;
  const int hl   = ln >> 4;        // lane half (0/1)
  const int lr   = ln & 15;        // lane within half
  const long rowbase = (long)blockIdx.x * 128 + wave * 16;
  const float* xrow = x + (rowbase + lr) * (long)EMB + hl * 8;

  v8f acc[3][4];
  #pragma unroll
  for (int m = 0; m < 3; ++m)
    #pragma unroll
    for (int t = 0; t < 4; ++t)
      acc[m][t] = {};

  // Per-thread W staging slots: 6 x v4f covers 3*64*32 floats per chunk.
  auto wload = [&](int ec, v4f wr[6]) {
    #pragma unroll
    for (int k2 = 0; k2 < 6; ++k2) {
      const int u = tid + k2 * 256;
      const int mat = u >> 9, r = u & 511, hrow = r >> 3, seg = r & 7;
      const float* Wm = (mat == 0) ? Wq : ((mat == 1) ? Wk : Wv);
      wr[k2] = *(const v4f*)(Wm + hrow * EMB + ec * 32 + seg * 4);
    }
  };
  auto wstore = [&](int buf, const v4f wr[6]) {
    _Float16* base = &ldsW[buf][0][0][0];
    #pragma unroll
    for (int k2 = 0; k2 < 6; ++k2) {
      const int u = tid + k2 * 256;
      const int mat = u >> 9, r = u & 511, hrow = r >> 3, seg = r & 7;
      v4h h;
      h[0] = (_Float16)wr[k2][0]; h[1] = (_Float16)wr[k2][1];
      h[2] = (_Float16)wr[k2][2]; h[3] = (_Float16)wr[k2][3];
      *(v4h*)(base + (mat * 64 + hrow) * 32 + seg * 4) = h;
    }
  };
  auto xload = [&](int ec, v4f xr[4]) {
    const float* xp = xrow + ec * 32;
    xr[0] = *(const v4f*)(xp + 0);
    xr[1] = *(const v4f*)(xp + 4);
    xr[2] = *(const v4f*)(xp + 16);
    xr[3] = *(const v4f*)(xp + 20);
  };

  // ---- prologue: stage chunk 0 ----
  v4f wr[6], xr[4];
  wload(0, wr);
  xload(0, xr);
  wstore(0, wr);
  v16h a = cvt_a(xr);
  __syncthreads();

  for (int ec = 0; ec < EMB / 32; ++ec) {
    const int  cur  = ec & 1;
    const bool more = (ec + 1) < EMB / 32;

    // Issue next chunk's global loads before compute (LOADcnt overlaps WMMAs).
    if (more) {
      wload(ec + 1, wr);
      xload(ec + 1, xr);
    }

    // 12 WMMAs; B fragments loaded in two alternating register groups of 4 so
    // both groups stay live and ds_load latency overlaps each WMMA burst.
    v16h bg0[4], bg1[4];
    #pragma unroll
    for (int t = 0; t < 4; ++t)
      bg0[t] = ld16_2(&ldsW[cur][0][t * 16 + lr][hl * 8], &ldsW[cur][0][t * 16 + lr][16 + hl * 8]);
    #pragma unroll
    for (int t = 0; t < 4; ++t)
      bg1[t] = ld16_2(&ldsW[cur][1][t * 16 + lr][hl * 8], &ldsW[cur][1][t * 16 + lr][16 + hl * 8]);
    #pragma unroll
    for (int t = 0; t < 4; ++t)
      acc[0][t] = wmma_f16(a, bg0[t], acc[0][t]);       // waits <= group 1 only
    #pragma unroll
    for (int t = 0; t < 4; ++t)
      bg0[t] = ld16_2(&ldsW[cur][2][t * 16 + lr][hl * 8], &ldsW[cur][2][t * 16 + lr][16 + hl * 8]);
    #pragma unroll
    for (int t = 0; t < 4; ++t)
      acc[1][t] = wmma_f16(a, bg1[t], acc[1][t]);       // overlaps group 2 loads
    #pragma unroll
    for (int t = 0; t < 4; ++t)
      acc[2][t] = wmma_f16(a, bg0[t], acc[2][t]);

    // Drain staged regs into the other LDS buffer, prep next A fragment.
    if (more) {
      wstore(1 - cur, wr);
      a = cvt_a(xr);
    }
    __syncthreads();
  }

  // Epilogue: + bias; Q pre-scaled by sqrt(HD)=8 (reference multiplies logits
  // by head_size**0.5); V stored transposed [B,64,S].
  #pragma unroll
  for (int t = 0; t < 4; ++t) {
    const int h = t * 16 + lr;
    const float bq_ = bq[h], bk_ = bk[h], bv_ = bv[h];
    #pragma unroll
    for (int r = 0; r < 8; ++r) {
      const long rowg = rowbase + r + hl * 8;         // C layout: M = r + 8*hl
      qbuf[rowg * HD + h] = (_Float16)((acc[0][t][r] + bq_) * 8.0f);
      kbuf[rowg * HD + h] = (_Float16)(acc[1][t][r] + bk_);
      const long b = rowg >> 11, s = rowg & (SEQ - 1);
      vtbuf[(b * HD + h) * SEQ + s] = (_Float16)(acc[2][t][r] + bv_);
    }
  }
}

// ---------------------------------------------------------------------------
// Stage 2: fused causal flash attention. grid = (S/64, B), 128 threads
// (4 waves, one 16-query tile each). 32-key blocks, double-buffered in LDS.
// ---------------------------------------------------------------------------
__global__ __launch_bounds__(128)
__attribute__((amdgpu_waves_per_eu(1, 4)))
void flash_attn_kernel(
    const _Float16* __restrict__ qbuf, const _Float16* __restrict__ kbuf,
    const _Float16* __restrict__ vtbuf, float* __restrict__ out)
{
  __shared__ __align__(32) _Float16 ldsK[2][32 * 64];   // [key][e]   2x4 KB
  __shared__ __align__(32) _Float16 ldsV[2][64 * 32];   // [h][key]   2x4 KB
  const int tid  = threadIdx.x;
  const int wave = tid >> 5;
  const int ln   = tid & 31;
  const int hl   = ln >> 4;
  const int lr   = ln & 15;
  const int batch  = blockIdx.y;
  const int qblock = blockIdx.x;            // 64 query rows per block
  const int qb     = qblock * 64 + wave * 16;
  const int qrow   = qb + lr;

  // Q^T B-fragments: n = query row (lane%16), k = head-dim e; two 8-half
  // runs per lane (k = 8*hl.., 16+8*hl..) from row-major Q. Loaded once, reused every iteration.
  const _Float16* qp = qbuf + ((long)batch * SEQ + qrow) * HD;
  const v16h bQ0 = ld16_2(qp + hl * 8, qp + 16 + hl * 8);
  const v16h bQ1 = ld16_2(qp + 32 + hl * 8, qp + 48 + hl * 8);

  // Per-thread staging slots: 2 K rows + 2 V rows (16B each) per 32-key block.
  const int krow0 = tid >> 3, kseg = tid & 7;     // K rows 0..15 / 16..31
  const int vrow0 = tid >> 2, vseg = tid & 3;     // V rows 0..31 / 32..63
  const _Float16* kbase = kbuf  + (long)batch * SEQ * HD;
  const _Float16* vbase = vtbuf + (long)batch * HD * SEQ;

  auto stage_load = [&](int kb, v4u s[4]) {
    s[0] = *(const v4u*)(kbase + (long)(kb + krow0) * HD + kseg * 8);
    s[1] = *(const v4u*)(kbase + (long)(kb + krow0 + 16) * HD + kseg * 8);
    s[2] = *(const v4u*)(vbase + (long)vrow0 * SEQ + kb + vseg * 8);
    s[3] = *(const v4u*)(vbase + (long)(vrow0 + 32) * SEQ + kb + vseg * 8);
  };
  auto stage_store = [&](int buf, const v4u s[4]) {
    *(v4u*)&ldsK[buf][krow0 * 64 + kseg * 8]        = s[0];
    *(v4u*)&ldsK[buf][(krow0 + 16) * 64 + kseg * 8] = s[1];
    *(v4u*)&ldsV[buf][vrow0 * 32 + vseg * 8]        = s[2];
    *(v4u*)&ldsV[buf][(vrow0 + 32) * 32 + vseg * 8] = s[3];
  };

  v8f O[4];                                  // O^T accum: rows = h, cols = q
  #pragma unroll
  for (int t = 0; t < 4; ++t) O[t] = {};
  float m = -1e30f, l = 0.0f;

  const int nj = 2 * (qblock + 1);           // causal: keys < (qblock+1)*64

  // ---- prologue: stage key-block 0 ----
  v4u sreg[4];
  stage_load(0, sreg);
  stage_store(0, sreg);
  __syncthreads();

  for (int j = 0; j < nj; ++j) {
    const int  cur  = j & 1;
    const bool more = (j + 1) < nj;
    if (more) stage_load((j + 1) * 32, sreg);   // global loads overlap compute

    // Pre-load ALL LDS fragments for this block: 4 K A-frags + 4 V A-frags.
    // V loads don't depend on softmax, so their latency hides behind the QK
    // WMMAs and the softmax VALU block.
    const _Float16* Kb = ldsK[cur];
    const _Float16* Vb = ldsV[cur];
    const _Float16* kp0 = &Kb[lr * 64];
    const _Float16* kp1 = &Kb[(16 + lr) * 64];
    v16h a00 = ld16_2(kp0 + hl * 8,      kp0 + 16 + hl * 8);
    v16h a01 = ld16_2(kp0 + 32 + hl * 8, kp0 + 48 + hl * 8);
    v16h a10 = ld16_2(kp1 + hl * 8,      kp1 + 16 + hl * 8);
    v16h a11 = ld16_2(kp1 + 32 + hl * 8, kp1 + 48 + hl * 8);
    v16h aV[4];
    #pragma unroll
    for (int t = 0; t < 4; ++t) {
      const _Float16* vp = &Vb[(t * 16 + lr) * 32];
      aV[t] = ld16(vp + hl * 16);
    }

    // S^T tiles: St[kt] = K_tile(16x64) @ Q^T(64x16)
    v8f St[2] = {{}, {}};
    St[0] = wmma_f16(a00, bQ0, St[0]);
    St[0] = wmma_f16(a01, bQ1, St[0]);
    St[1] = wmma_f16(a10, bQ0, St[1]);
    St[1] = wmma_f16(a11, bQ1, St[1]);

    // Causal mask (scale already folded into Q) + online softmax. Query index
    // is the lane axis -> row stats need a single xor-16 reduction.
    const int kb = j * 32;
    float mx = -1e30f;
    #pragma unroll
    for (int r = 0; r < 8; ++r) {
      const int key0 = kb + r + hl * 8;       // St[kt] row = key within tile
      if (key0 > qrow)      St[0][r] = -1e30f;
      if (key0 + 16 > qrow) St[1][r] = -1e30f;
      mx = fmaxf(mx, fmaxf(St[0][r], St[1][r]));
    }
    mx = fmaxf(mx, __shfl_xor(mx, 16));
    const float mnew  = fmaxf(m, mx);
    const float alpha = __expf(m - mnew);
    m = mnew;

    float p0[8], p1[8], rs = 0.0f;
    #pragma unroll
    for (int r = 0; r < 8; ++r) {
      p0[r] = __expf(St[0][r] - mnew);
      p1[r] = __expf(St[1][r] - mnew);
      rs += p0[r] + p1[r];
    }
    rs += __shfl_xor(rs, 16);
    l = l * alpha + rs;

    #pragma unroll
    for (int t = 0; t < 4; ++t)
      #pragma unroll
      for (int r = 0; r < 8; ++r)
        O[t][r] *= alpha;

    // Build P^T B-fragment (32 keys x 16 queries) with one half-wave swap:
    // lanes 0-15 need keys 0-15 (own St0 rows 0-7, partner's rows 8-15);
    // lanes 16-31 need keys 16-31 (partner's St1 rows 0-7, own rows 8-15).
    v16h bP;
    const bool lo = (hl == 0);
    #pragma unroll
    for (int r = 0; r < 8; ++r) {
      const float o0 = __shfl_xor(p0[r], 16);
      const float o1 = __shfl_xor(p1[r], 16);
      bP[r]     = (_Float16)(lo ? p0[r] : o1);
      bP[8 + r] = (_Float16)(lo ? o0 : p1[r]);
    }

    // O^T += V^T_tile(16x32) @ P^T(32x16) per 16-wide head-dim tile.
    #pragma unroll
    for (int t = 0; t < 4; ++t)
      O[t] = wmma_f16(aV[t], bP, O[t]);

    // Drain staged next block into the other buffer; single barrier per iter.
    if (more) stage_store(1 - cur, sreg);
    __syncthreads();
  }

  // Normalize and store: out[b][qrow][h], h = 16t + r + 8*hl, qrow = lane col.
  const float inv = 1.0f / l;
  float* op = out + ((long)batch * SEQ + qrow) * HD;
  #pragma unroll
  for (int t = 0; t < 4; ++t)
    #pragma unroll
    for (int r = 0; r < 8; ++r)
      op[t * 16 + r + hl * 8] = O[t][r] * inv;
}

// ---------------------------------------------------------------------------
extern "C" void kernel_launch(void* const* d_in, const int* in_sizes, int n_in,
                              void* d_out, int out_size, void* d_ws, size_t ws_size,
                              hipStream_t stream) {
  (void)in_sizes; (void)n_in; (void)out_size; (void)ws_size;
  const float* x  = (const float*)d_in[0];
  const float* Wk = (const float*)d_in[1];
  const float* bk = (const float*)d_in[2];
  const float* Wq = (const float*)d_in[3];
  const float* bq = (const float*)d_in[4];
  const float* Wv = (const float*)d_in[5];
  const float* bv = (const float*)d_in[6];
  float* out = (float*)d_out;

  const size_t n = (size_t)BATCH * SEQ * HD;     // 1M elems -> 2 MB each (f16)
  _Float16* qbuf  = (_Float16*)d_ws;
  _Float16* kbuf  = qbuf + n;
  _Float16* vtbuf = kbuf + n;                    // total 6 MB of workspace

  qkv_proj_kernel<<<dim3((BATCH * SEQ) / 128), dim3(256), 0, stream>>>(
      x, Wk, bk, Wq, bq, Wv, bv, qbuf, kbuf, vtbuf);
  flash_attn_kernel<<<dim3(SEQ / 64, BATCH), dim3(128), 0, stream>>>(
      qbuf, kbuf, vtbuf, out);
}
